// FlatConv3x3NNUE_47519518163393
// MI455X (gfx1250) — matrix operand order repaired and checked by reading, not compiled, not run on hardware
//
#include <hip/hip_runtime.h>
#include <hip/hip_bf16.h>
#include <math.h>

typedef __attribute__((ext_vector_type(16))) _Float16 v16h;
typedef __attribute__((ext_vector_type(8)))  float    v8f;

#define B_N   4096
#define T_N   25
#define QMAXV (127.0f/128.0f)
#define EPSBN 1e-5f

// ---------------- workspace layout (float offsets) ----------------
#define OFF_SUM1 0
#define OFF_SQ1  3200
#define OFF_SUM2 6400
#define OFF_SQ2  9600
#define OFF_SUM3 12800
#define OFF_SQ3  16000
#define OFF_SUM4 19200
#define OFF_SQ4  20000
#define OFF_FEAT 20800
#define ZERO_N   (20800 + B_N*32)      // stats + feat get zeroed each launch
#define OFF_SC1  151872
#define OFF_SH1  155072
#define OFF_SC2  158272
#define OFF_SH2  161472
#define OFF_SC3  164672
#define OFF_SH3  167872
#define OFF_SC4  171072
#define OFF_SH4  171872
#define F32_END  172672                // halves region starts here (32B aligned)
#define W2F_HALVES (T_N*16*8*32*16)    // 1,638,400 halves
#define W3F_HALVES (T_N*4*8*32*16)     //   409,600 halves
#define W4F_HALVES (T_N*4*2*32*16)     //   102,400 halves

// ---------------- LDS layout for tower kernels (bytes) ----------------
#define LDS_W2S   0            // 131072  (per-tower W2 fragments, f16)
#define LDS_W3S   131072       // 32768
#define LDS_W4S   163840       // 8192
#define LDS_A2S   172032       // 32768  (8 waves x 16x128 f16 staging)
#define LDS_XS    204800       // 50176  (128 rows x 98 f32 of x)
#define LDS_W1S   254976       // 4096   (W1[t], f32)
#define LDS_SC    259072       // 3328   (folded BN scale/shift)
#define LDS_STATS 262400       // 1024   (block sum[128], sq[128])
#define TOWER_SMEM 263424
#define STATS1_SMEM (50176 + 4096 + 2048)

__device__ __forceinline__ float mishf(float z) {
  float sp = (z > 20.f) ? z : log1pf(__expf(z));
  return z * tanhf(sp);
}
__device__ __forceinline__ float fq128(float w) {
  return fminf(fmaxf(rintf(w * 128.f), -128.f), 127.f) * (1.f/128.f);
}
__device__ __forceinline__ float fqbias(float b) {
  return rintf(b * 16384.f) * (1.f/16384.f);
}
__device__ __forceinline__ v8f vzero() {
  v8f r;
  #pragma unroll
  for (int e = 0; e < 8; ++e) r[e] = 0.f;
  return r;
}

// ---------------- zero stats + feat ----------------
__global__ void zero_kernel(float* __restrict__ ws) {
  int i = blockIdx.x * 256 + threadIdx.x;
  if (i < ZERO_N) ws[i] = 0.f;
}

// ---------------- pre-swizzle W2/W3/W4 into WMMA B-fragment order ----------
// B-fragment order used here (contiguous k): lane L holds N = L&15; half j holds
// K = j + 16*(L>>4).  Fragment order: (((t*S + s)*NT + nt)*32 + lane)*16.
__global__ __launch_bounds__(256) void prep_kernel(const float* __restrict__ W2,
                                                   const float* __restrict__ W3,
                                                   const float* __restrict__ W4,
                                                   float* __restrict__ ws) {
  _Float16* w2f = (_Float16*)(ws + F32_END);
  _Float16* w3f = w2f + W2F_HALVES;
  _Float16* w4f = w3f + W3F_HALVES;
  int slot = blockIdx.x * 256 + threadIdx.x;
  if (slot < 102400) {                       // W2: t(25) s(16) nt(8) lane(32)
    int lane = slot & 31; int rem = slot >> 5;
    int nt = rem & 7; rem >>= 3; int s = rem & 15; int t = rem >> 4;
    int gg = lane >> 4, n = lane & 15;
    int d = nt * 16 + n;
    _Float16* dst = w2f + (size_t)slot * 16;
    #pragma unroll
    for (int j = 0; j < 16; ++j) {
      int kk = s * 32 + j + 16 * gg;         // kappa = c*4 + p
      int c = kk >> 2, p = kk & 3;
      dst[j] = (_Float16)W2[((size_t)(t * 128 + d) * 128 + c) * 4 + p];
    }
  } else if (slot < 128000) {                // W3: t(25) s(4) nt(8) lane(32)
    int sl = slot - 102400;
    int lane = sl & 31; int rem = sl >> 5;
    int nt = rem & 7; rem >>= 3; int s = rem & 3; int t = rem >> 2;
    int gg = lane >> 4, n = lane & 15;
    int d = nt * 16 + n;
    _Float16* dst = w3f + (size_t)sl * 16;
    #pragma unroll
    for (int j = 0; j < 16; ++j) {
      int c = s * 32 + j + 16 * gg;
      dst[j] = (_Float16)W3[(size_t)(t * 128 + d) * 128 + c];
    }
  } else if (slot < 134400) {                // W4: t(25) s(4) nt(2) lane(32)
    int sl = slot - 128000;
    int lane = sl & 31; int rem = sl >> 5;
    int nt = rem & 1; rem >>= 1; int s = rem & 3; int t = rem >> 2;
    int gg = lane >> 4, n = lane & 15;
    int f = nt * 16 + n;
    _Float16* dst = w4f + (size_t)sl * 16;
    #pragma unroll
    for (int j = 0; j < 16; ++j) {
      int c = s * 32 + j + 16 * gg;
      dst[j] = (_Float16)W4[(size_t)(t * 32 + f) * 128 + c];
    }
  }
}

// ---------------- stats of raw conv1 output (BN1 over axes (B,p)) ----------
__global__ __launch_bounds__(256) void stats1_kernel(const float* __restrict__ x,
                                                     const float* __restrict__ W1,
                                                     float* __restrict__ ws) {
  extern __shared__ char smem[];
  float* xs  = (float*)smem;                 // 128*98
  float* W1s = xs + 128 * 98;                // 1024
  float* red = W1s + 1024;                   // 512
  const int tid = threadIdx.x, t = blockIdx.x, rowBase = blockIdx.y * 128;
  const int ty = t / 5, tx = t % 5;
  for (int k = tid; k < 12544; k += 256) xs[k] = x[(size_t)rowBase * 98 + k];
  for (int k = tid; k < 1024;  k += 256) W1s[k] = W1[t * 1024 + k];
  __syncthreads();
  const int c = tid & 127, rh = tid >> 7;
  const float* w = W1s + c * 8;
  float su = 0.f, sq = 0.f;
  for (int rl = 0; rl < 64; ++rl) {
    const float* xr = xs + (rh * 64 + rl) * 98;
    #pragma unroll
    for (int p = 0; p < 4; ++p) {
      const float* xb = xr + (ty + (p >> 1)) * 7 + (tx + (p & 1));
      float h = xb[0]*w[0] + xb[1]*w[1] + xb[7]*w[2] + xb[8]*w[3]
              + xb[49]*w[4] + xb[50]*w[5] + xb[56]*w[6] + xb[57]*w[7];
      su += h; sq += h * h;
    }
  }
  red[tid] = su; red[256 + tid] = sq;
  __syncthreads();
  if (tid < 128) {
    atomicAdd(&ws[OFF_SUM1 + t * 128 + tid], red[tid] + red[tid + 128]);
    atomicAdd(&ws[OFF_SQ1  + t * 128 + tid], red[256 + tid] + red[256 + tid + 128]);
  }
}

// ---------------- fold BN into scale/shift ----------------
__global__ void finalize_kernel(float* __restrict__ ws, const float* __restrict__ gg,
                                const float* __restrict__ bb, int sumOff, int sqOff,
                                int scOff, int shOff, int nElem, float invCount) {
  int i = blockIdx.x * 256 + threadIdx.x;
  if (i >= nElem) return;
  float mean = ws[sumOff + i] * invCount;
  float var  = ws[sqOff + i] * invCount - mean * mean;
  float rs   = rsqrtf(var + EPSBN);
  float sc   = gg[i] * rs;
  ws[scOff + i] = sc;
  ws[shOff + i] = bb[i] - mean * sc;
}

// ---------------- fused tower pipeline ----------------
// STAGE==2: conv1->bn1->mish->conv2 (WMMA), emit stats2
// STAGE==3: ...->bn2->mish->conv3 (WMMA), emit stats3
// STAGE==4: ...->bn3->mish->conv4 (WMMA), emit stats4
// STAGE==5: ...->bn4->clip->fq, atomic-accumulate feat over towers
template <int STAGE>
__global__ __launch_bounds__(256) void tower_kernel(const float* __restrict__ x,
                                                    const float* __restrict__ W1,
                                                    float* __restrict__ ws) {
  extern __shared__ char smem[];
  _Float16* W2s = (_Float16*)(smem + LDS_W2S);
  _Float16* W3s = (_Float16*)(smem + LDS_W3S);
  _Float16* W4s = (_Float16*)(smem + LDS_W4S);
  _Float16* a2s = (_Float16*)(smem + LDS_A2S);
  float* xs   = (float*)(smem + LDS_XS);
  float* W1s  = (float*)(smem + LDS_W1S);
  float* scs  = (float*)(smem + LDS_SC);
  float* stat = (float*)(smem + LDS_STATS);   // [0..127]=sum, [128..255]=sumsq

  const _Float16* w2f = (const _Float16*)(ws + F32_END);
  const _Float16* w3f = w2f + W2F_HALVES;
  const _Float16* w4f = w3f + W3F_HALVES;

  const int tid = threadIdx.x;
  const int t = blockIdx.x;
  const int rowBase = blockIdx.y * 128;
  const int ty = t / 5, tx = t % 5;

  // cooperative staging
  for (int k = tid; k < 12544; k += 256) xs[k] = x[(size_t)rowBase * 98 + k];
  for (int k = tid; k < 1024;  k += 256) W1s[k] = W1[t * 1024 + k];
  if (tid < 128) {
    scs[tid]       = ws[OFF_SC1 + t * 128 + tid];
    scs[128 + tid] = ws[OFF_SH1 + t * 128 + tid];
    if (STAGE >= 3) { scs[256 + tid] = ws[OFF_SC2 + t * 128 + tid];
                      scs[384 + tid] = ws[OFF_SH2 + t * 128 + tid]; }
    if (STAGE >= 4) { scs[512 + tid] = ws[OFF_SC3 + t * 128 + tid];
                      scs[640 + tid] = ws[OFF_SH3 + t * 128 + tid]; }
    if (STAGE >= 5 && tid < 32) { scs[768 + tid] = ws[OFF_SC4 + t * 32 + tid];
                                  scs[800 + tid] = ws[OFF_SH4 + t * 32 + tid]; }
  }
  { const uint4* s4 = (const uint4*)(w2f + (size_t)t * 65536); uint4* d4 = (uint4*)W2s;
    for (int k = tid; k < 8192; k += 256) d4[k] = s4[k]; }
  if (STAGE >= 3) {
    const uint4* s4 = (const uint4*)(w3f + (size_t)t * 16384); uint4* d4 = (uint4*)W3s;
    for (int k = tid; k < 2048; k += 256) d4[k] = s4[k]; }
  if (STAGE >= 4) {
    const uint4* s4 = (const uint4*)(w4f + (size_t)t * 4096); uint4* d4 = (uint4*)W4s;
    for (int k = tid; k < 512; k += 256) d4[k] = s4[k]; }
  if (STAGE <= 4) stat[tid] = 0.f;
  __syncthreads();

  const int wave = tid >> 5, lane = tid & 31;
  const int g = lane >> 4, n = lane & 15;
  const int g8 = g * 8;
  const float* xrow = xs + (wave * 16 + n) * 98;   // A-frag row m = lane&15
  const float* sc1 = scs, *sh1 = scs + 128;

  // ---- conv2: [16 x 512] x [512 x 128], K-steps of 32 ----
  v8f acc2[8];
  #pragma unroll
  for (int i = 0; i < 8; ++i) acc2[i] = vzero();
  for (int s = 0; s < 16; ++s) {
    v16h A;
    #pragma unroll
    for (int j = 0; j < 16; ++j) {
      // A order used here (contiguous k, same as B prep): half j holds K = j + 16*(lane>>4)
      const int kk = s * 32 + j + 16 * g;
      const int c = kk >> 2, p = kk & 3;
      const float* xb = xrow + (ty + (p >> 1)) * 7 + (tx + (p & 1));
      const float* w  = W1s + c * 8;
      float h = xb[0]*w[0] + xb[1]*w[1] + xb[7]*w[2] + xb[8]*w[3]
              + xb[49]*w[4] + xb[50]*w[5] + xb[56]*w[6] + xb[57]*w[7];
      A[j] = (_Float16)mishf(h * sc1[c] + sh1[c]);
    }
    const _Float16* bb = W2s + ((size_t)(s * 256) + lane) * 16;
    #pragma unroll
    for (int nt = 0; nt < 8; ++nt) {
      v16h Bf = *(const v16h*)(bb + nt * 512);
      acc2[nt] = __builtin_amdgcn_wmma_f32_16x16x32_f16(false, A, false, Bf,
                                                        (short)0, acc2[nt], false, false);
    }
  }

  if (STAGE == 2) {
    #pragma unroll
    for (int nt = 0; nt < 8; ++nt) {
      float su = 0.f, sq = 0.f;
      #pragma unroll
      for (int r = 0; r < 8; ++r) { float v = acc2[nt][r]; su += v; sq += v * v; }
      atomicAdd(&stat[nt * 16 + n], su);
      atomicAdd(&stat[128 + nt * 16 + n], sq);
    }
    __syncthreads();
    if (tid < 128) {
      atomicAdd(&ws[OFF_SUM2 + t * 128 + tid], stat[tid]);
      atomicAdd(&ws[OFF_SQ2  + t * 128 + tid], stat[128 + tid]);
    }
    return;
  }

  // ---- bn2 + mish, stage a2 (per-wave private 16x128 tile) ----
  _Float16* aw = a2s + wave * 2048;
  {
    const float* sc2 = scs + 256, *sh2 = scs + 384;
    #pragma unroll
    for (int nt = 0; nt < 8; ++nt) {
      const int d = nt * 16 + n;
      const float a = sc2[d], b = sh2[d];
      #pragma unroll
      for (int r = 0; r < 8; ++r)                 // D layout: row m = r + 8g
        aw[(r + g8) * 128 + d] = (_Float16)mishf(acc2[nt][r] * a + b);
    }
  }

  // ---- conv3: [16 x 128] x [128 x 128] ----
  v8f acc3[8];
  #pragma unroll
  for (int i = 0; i < 8; ++i) acc3[i] = vzero();
  for (int s = 0; s < 4; ++s) {
    v16h A;
    #pragma unroll
    for (int j = 0; j < 16; ++j) {
      const int kk = s * 32 + j + 16 * g;
      A[j] = aw[n * 128 + kk];
    }
    const _Float16* bb = W3s + ((size_t)(s * 256) + lane) * 16;
    #pragma unroll
    for (int nt = 0; nt < 8; ++nt) {
      v16h Bf = *(const v16h*)(bb + nt * 512);
      acc3[nt] = __builtin_amdgcn_wmma_f32_16x16x32_f16(false, A, false, Bf,
                                                        (short)0, acc3[nt], false, false);
    }
  }

  if (STAGE == 3) {
    #pragma unroll
    for (int nt = 0; nt < 8; ++nt) {
      float su = 0.f, sq = 0.f;
      #pragma unroll
      for (int r = 0; r < 8; ++r) { float v = acc3[nt][r]; su += v; sq += v * v; }
      atomicAdd(&stat[nt * 16 + n], su);
      atomicAdd(&stat[128 + nt * 16 + n], sq);
    }
    __syncthreads();
    if (tid < 128) {
      atomicAdd(&ws[OFF_SUM3 + t * 128 + tid], stat[tid]);
      atomicAdd(&ws[OFF_SQ3  + t * 128 + tid], stat[128 + tid]);
    }
    return;
  }

  // ---- bn3 + mish, stage a3 (reuse per-wave tile) ----
  {
    const float* sc3 = scs + 512, *sh3 = scs + 640;
    #pragma unroll
    for (int nt = 0; nt < 8; ++nt) {
      const int d = nt * 16 + n;
      const float a = sc3[d], b = sh3[d];
      #pragma unroll
      for (int r = 0; r < 8; ++r)
        aw[(r + g8) * 128 + d] = (_Float16)mishf(acc3[nt][r] * a + b);
    }
  }

  // ---- conv4: [16 x 128] x [128 x 32] ----
  v8f acc4[2];
  #pragma unroll
  for (int i = 0; i < 2; ++i) acc4[i] = vzero();
  for (int s = 0; s < 4; ++s) {
    v16h A;
    #pragma unroll
    for (int j = 0; j < 16; ++j) {
      const int kk = s * 32 + j + 16 * g;
      A[j] = aw[n * 128 + kk];
    }
    const _Float16* bb = W4s + ((size_t)(s * 64) + lane) * 16;
    #pragma unroll
    for (int nt = 0; nt < 2; ++nt) {
      v16h Bf = *(const v16h*)(bb + nt * 512);
      acc4[nt] = __builtin_amdgcn_wmma_f32_16x16x32_f16(false, A, false, Bf,
                                                        (short)0, acc4[nt], false, false);
    }
  }

  if (STAGE == 4) {
    #pragma unroll
    for (int nt = 0; nt < 2; ++nt) {
      float su = 0.f, sq = 0.f;
      #pragma unroll
      for (int r = 0; r < 8; ++r) { float v = acc4[nt][r]; su += v; sq += v * v; }
      atomicAdd(&stat[nt * 16 + n], su);
      atomicAdd(&stat[128 + nt * 16 + n], sq);
    }
    __syncthreads();
    if (tid < 32) {
      atomicAdd(&ws[OFF_SUM4 + t * 32 + tid], stat[tid]);
      atomicAdd(&ws[OFF_SQ4  + t * 32 + tid], stat[128 + tid]);
    }
    return;
  }

  // ---- STAGE 5: bn4, clip, fake-int8 quant, accumulate feat over towers ----
  {
    const float* sc4 = scs + 768, *sh4 = scs + 800;
    float* feat = ws + OFF_FEAT;
    #pragma unroll
    for (int nt = 0; nt < 2; ++nt) {
      const int f = nt * 16 + n;
      const float a = sc4[f], bsh = sh4[f];
      #pragma unroll
      for (int r = 0; r < 8; ++r) {
        float h = acc4[nt][r] * a + bsh;
        float v = fminf(fmaxf(h, -1.f), QMAXV);
        float q = fq128(v);
        const int b = rowBase + wave * 16 + r + g8;
        atomicAdd(&feat[b * 32 + f], q);
      }
    }
  }
}

// ---------------- head MLP + policy zeros ----------------
__global__ __launch_bounds__(256) void head_kernel(const float* __restrict__ ws,
                                                   const float* __restrict__ Wf1,
                                                   const float* __restrict__ bf1,
                                                   const float* __restrict__ Wf2,
                                                   const float* __restrict__ bf2,
                                                   const float* __restrict__ Wf3,
                                                   const float* __restrict__ bf3,
                                                   float* __restrict__ out) {
  int b = blockIdx.x * 256 + threadIdx.x;
  if (b >= B_N) return;
  const float* feat = ws + OFF_FEAT + b * 32;
  float u[32], h1[32], h2[32];
  for (int i = 0; i < 32; ++i) u[i] = fminf(fmaxf(feat[i], -1.f), QMAXV);
  for (int o = 0; o < 32; ++o) {
    float a = fqbias(bf1[o]);
    for (int i = 0; i < 32; ++i) a += u[i] * fq128(Wf1[o * 32 + i]);
    h1[o] = fminf(fmaxf(a, 0.f), QMAXV);
  }
  for (int o = 0; o < 32; ++o) {
    float a = fqbias(bf2[o]);
    for (int i = 0; i < 32; ++i) a += h1[i] * fq128(Wf2[o * 32 + i]);
    h2[o] = fminf(fmaxf(a, 0.f), QMAXV);
  }
  for (int o = 0; o < 3; ++o) {
    float a = fqbias(bf3[o]);
    for (int i = 0; i < 32; ++i) a += h2[i] * fq128(Wf3[o * 32 + i]);
    out[b * 3 + o] = a;
  }
  float* pol = out + (size_t)B_N * 3 + (size_t)b * 49;
  for (int k = 0; k < 49; ++k) pol[k] = 0.f;
}

extern "C" void kernel_launch(void* const* d_in, const int* in_sizes, int n_in,
                              void* d_out, int out_size, void* d_ws, size_t ws_size,
                              hipStream_t stream) {
  (void)in_sizes; (void)n_in; (void)out_size; (void)ws_size;
  const float* x   = (const float*)d_in[0];
  const float* W1  = (const float*)d_in[1];
  const float* g1  = (const float*)d_in[2];  const float* b1  = (const float*)d_in[3];
  const float* W2  = (const float*)d_in[4];
  const float* g2  = (const float*)d_in[5];  const float* b2  = (const float*)d_in[6];
  const float* W3  = (const float*)d_in[7];
  const float* g3  = (const float*)d_in[8];  const float* b3  = (const float*)d_in[9];
  const float* W4  = (const float*)d_in[10];
  const float* g4  = (const float*)d_in[11]; const float* b4  = (const float*)d_in[12];
  const float* Wf1 = (const float*)d_in[13]; const float* bf1 = (const float*)d_in[14];
  const float* Wf2 = (const float*)d_in[15]; const float* bf2 = (const float*)d_in[16];
  const float* Wf3 = (const float*)d_in[17]; const float* bf3 = (const float*)d_in[18];
  float* ws  = (float*)d_ws;
  float* out = (float*)d_out;

  // allow >64KB dynamic LDS on the tower kernels (WGP supports 320KB/WG)
  (void)hipFuncSetAttribute((const void*)&tower_kernel<2>,
                            hipFuncAttributeMaxDynamicSharedMemorySize, TOWER_SMEM);
  (void)hipFuncSetAttribute((const void*)&tower_kernel<3>,
                            hipFuncAttributeMaxDynamicSharedMemorySize, TOWER_SMEM);
  (void)hipFuncSetAttribute((const void*)&tower_kernel<4>,
                            hipFuncAttributeMaxDynamicSharedMemorySize, TOWER_SMEM);
  (void)hipFuncSetAttribute((const void*)&tower_kernel<5>,
                            hipFuncAttributeMaxDynamicSharedMemorySize, TOWER_SMEM);

  dim3 grid(T_N, B_N / 128);

  zero_kernel<<<(ZERO_N + 255) / 256, 256, 0, stream>>>(ws);
  prep_kernel<<<525, 256, 0, stream>>>(W2, W3, W4, ws);

  stats1_kernel<<<grid, 256, STATS1_SMEM, stream>>>(x, W1, ws);
  finalize_kernel<<<13, 256, 0, stream>>>(ws, g1, b1, OFF_SUM1, OFF_SQ1,
                                          OFF_SC1, OFF_SH1, 3200, 1.f / (B_N * 4.f));

  tower_kernel<2><<<grid, 256, TOWER_SMEM, stream>>>(x, W1, ws);
  finalize_kernel<<<13, 256, 0, stream>>>(ws, g2, b2, OFF_SUM2, OFF_SQ2,
                                          OFF_SC2, OFF_SH2, 3200, 1.f / (float)B_N);

  tower_kernel<3><<<grid, 256, TOWER_SMEM, stream>>>(x, W1, ws);
  finalize_kernel<<<13, 256, 0, stream>>>(ws, g3, b3, OFF_SUM3, OFF_SQ3,
                                          OFF_SC3, OFF_SH3, 3200, 1.f / (float)B_N);

  tower_kernel<4><<<grid, 256, TOWER_SMEM, stream>>>(x, W1, ws);
  finalize_kernel<<<4, 256, 0, stream>>>(ws, g4, b4, OFF_SUM4, OFF_SQ4,
                                         OFF_SC4, OFF_SH4, 800, 1.f / (float)B_N);

  tower_kernel<5><<<grid, 256, TOWER_SMEM, stream>>>(x, W1, ws);

  head_kernel<<<16, 256, 0, stream>>>(ws, Wf1, bf1, Wf2, bf2, Wf3, bf3, out);
}
